// EMAVQQuantizer_352187318812
// MI455X (gfx1250) — matrix operand order repaired and checked by reading, not compiled, not run on hardware
//
#include <hip/hip_runtime.h>
#include <hip/hip_bf16.h>

typedef __attribute__((ext_vector_type(16))) _Float16 v16h;
typedef __attribute__((ext_vector_type(8)))  float    v8f;
typedef __attribute__((ext_vector_type(4)))  unsigned int u32x4;
typedef __attribute__((ext_vector_type(8)))  int      i32x8;
typedef __attribute__((ext_vector_type(4)))  int      i32x4;

#define D 256
#define K 4096
#define NROWS 65536              // 16*4096
#define NELEM 16777216           // NROWS * D
#define TILE_BYTES (16 * D * 2)  // 16 codes x 256 halves = 8192 B, contiguous

// ---------------------------------------------------------------------------
// Kernel 1: embedding fp32 -> f16 copy + per-code squared norms
// ---------------------------------------------------------------------------
__global__ __launch_bounds__(256) void vq_prep_kernel(
    const float* __restrict__ emb, _Float16* __restrict__ embF16,
    float* __restrict__ enorm)
{
    const int k = blockIdx.x;
    const int d = threadIdx.x;
    const float v = emb[(size_t)k * D + d];
    embF16[(size_t)k * D + d] = (_Float16)v;
    __shared__ float s[256];
    s[d] = v * v;
    __syncthreads();
    #pragma unroll
    for (int o = 128; o > 0; o >>= 1) {
        if (d < o) s[d] += s[d + o];
        __syncthreads();
    }
    if (d == 0) enorm[k] = s[0];
}

// ---------------------------------------------------------------------------
// Kernel 2: WMMA distance GEMM + fused argmin, TDM-fed double-buffered LDS.
// 8 waves/WG, each wave owns 16 rows (A register-resident, 8 x v16h).
// One tensor_load_to_lds per 16-code tile per WG, shared by all 8 waves.
// blockDim = 256, grid = NROWS/128 = 512.
// ---------------------------------------------------------------------------
__device__ __forceinline__ void tdm_issue_tile(
    unsigned long long gaddr, unsigned ldsAddr)
{
    // D# group 0: count=1 | lds_addr | global_addr | type=2
    u32x4 g0;
    g0[0] = 1u;
    g0[1] = ldsAddr;
    g0[2] = (unsigned)(gaddr & 0xFFFFFFFFull);
    g0[3] = (unsigned)((gaddr >> 32) & 0x1FFFFFFull) | (2u << 30);
    // D# group 1: data_size=2B; tensor_dim0=tile_dim0=4096 elems; dim1=1
    i32x8 g1;
    g1[0] = 0x00010000;                  // data_size=1 (2 bytes/elem)
    g1[1] = (int)(4096u << 16);          // tensor_dim0[15:0]
    g1[2] = (int)(1u << 16);             // tensor_dim0[31:16]=0 | tensor_dim1=1
    g1[3] = (int)(4096u << 16);          // tile_dim0=4096
    g1[4] = 1;                           // tile_dim1=1, tile_dim2=0
    g1[5] = 4096;                        // tensor_dim0_stride[31:0]
    g1[6] = (int)(4096u << 16);          // stride0[47:32]=0 | stride1[15:0]
    g1[7] = 0;
    i32x4 z4 = {0, 0, 0, 0};
    i32x8 z8 = {0, 0, 0, 0, 0, 0, 0, 0};
    __builtin_amdgcn_tensor_load_to_lds(g0, g1, z4, z4, z8, 0);
}

__global__ __launch_bounds__(256) void vq_argmin_kernel(
    const float* __restrict__ z_e, const _Float16* __restrict__ embF16,
    const float* __restrict__ enorm, int* __restrict__ codeI,
    float* __restrict__ codeF)
{
    __shared__ alignas(64) _Float16 ldsB[2][16 * D];   // 2 x 8 KB double buffer

    const int lane    = threadIdx.x & 31;
    const int wave    = threadIdx.x >> 5;
    const int rowBase = (blockIdx.x * 8 + wave) * 16;
    const int col     = lane & 15;     // N within tile / A row M
    const int grp     = lane >> 4;     // half-wave selector

    // ---- Load A: row M = col; per ISA 16-bit A 16x32 layout:
    //      half h -> K = chunk*32 + grp*8 + (h<8 ? h : 16 + (h-8))
    v16h a[8];
    const float* zr = z_e + (size_t)(rowBase + col) * D;
    #pragma unroll
    for (int ch = 0; ch < 8; ++ch) {
        const int base = ch * 32 + grp * 8;
        const float4 f0 = *(const float4*)(zr + base + 0);
        const float4 f1 = *(const float4*)(zr + base + 4);
        const float4 f2 = *(const float4*)(zr + base + 16);
        const float4 f3 = *(const float4*)(zr + base + 20);
        v16h t;
        t[0]  = (_Float16)f0.x; t[1]  = (_Float16)f0.y;
        t[2]  = (_Float16)f0.z; t[3]  = (_Float16)f0.w;
        t[4]  = (_Float16)f1.x; t[5]  = (_Float16)f1.y;
        t[6]  = (_Float16)f1.z; t[7]  = (_Float16)f1.w;
        t[8]  = (_Float16)f2.x; t[9]  = (_Float16)f2.y;
        t[10] = (_Float16)f2.z; t[11] = (_Float16)f2.w;
        t[12] = (_Float16)f3.x; t[13] = (_Float16)f3.y;
        t[14] = (_Float16)f3.z; t[15] = (_Float16)f3.w;
        a[ch] = t;
    }

    const unsigned ldsAddr0 = (unsigned)(uintptr_t)&ldsB[0][0];
    const unsigned ldsAddr1 = (unsigned)(uintptr_t)&ldsB[1][0];
    const unsigned long long gbase = (unsigned long long)(uintptr_t)embF16;

    // Prime the pipeline: tile 0 -> buffer 0.
    if (wave == 0) {
        tdm_issue_tile(gbase, ldsAddr0);
        __builtin_amdgcn_s_wait_tensorcnt(0);
    }
    __syncthreads();

    float minv[8];
    int   mina[8];
    #pragma unroll
    for (int j = 0; j < 8; ++j) { minv[j] = 3.4e38f; mina[j] = 0; }

    for (int ct = 0; ct < K / 16; ++ct) {
        // Prefetch next tile into the alternate buffer (drained last iter).
        if (wave == 0 && ct + 1 < K / 16) {
            tdm_issue_tile(gbase + (unsigned long long)(ct + 1) * TILE_BYTES,
                           (ct & 1) ? ldsAddr0 : ldsAddr1);
        }

        // B layout: half h -> K = chunk*32 + grp*8 + (h<8 ? h : 16 + (h-8)), column N = col
        const _Float16* bb = &ldsB[ct & 1][0] + col * D + grp * 8;
        v8f acc = {};
        #pragma unroll
        for (int ch = 0; ch < 8; ++ch) {
            typedef __attribute__((ext_vector_type(8))) _Float16 v8h_t;
            const v8h_t b0 = *(const v8h_t*)(bb + ch * 32);
            const v8h_t b1 = *(const v8h_t*)(bb + ch * 32 + 16);
            v16h b = __builtin_shufflevector(b0, b1, 0, 1, 2, 3, 4, 5, 6, 7, 8, 9, 10, 11, 12, 13, 14, 15);
            acc = __builtin_amdgcn_wmma_f32_16x16x32_f16(
                false, a[ch], false, b, (short)0, acc, false, false);
        }

        const int   c  = ct * 16 + col;
        const float en = enorm[c];
        // C layout: VGPR j holds row (j + 8*grp), column = col
        #pragma unroll
        for (int j = 0; j < 8; ++j) {
            const float dist = en - 2.0f * acc[j];
            if (dist < minv[j]) { minv[j] = dist; mina[j] = c; }
        }

        // Publish the prefetched tile to all waves.
        if (wave == 0) __builtin_amdgcn_s_wait_tensorcnt(0);
        __syncthreads();
    }

    // Min-reduce across the 16 columns (lanes within each half-wave).
    #pragma unroll
    for (int off = 1; off < 16; off <<= 1) {
        #pragma unroll
        for (int j = 0; j < 8; ++j) {
            const float om = __shfl_xor(minv[j], off, 32);
            const int   oa = __shfl_xor(mina[j], off, 32);
            if (om < minv[j] || (om == minv[j] && oa < mina[j])) {
                minv[j] = om; mina[j] = oa;
            }
        }
    }
    if (col == 0) {
        #pragma unroll
        for (int j = 0; j < 8; ++j) {
            const int r = rowBase + grp * 8 + j;
            codeI[r] = mina[j];
            codeF[r] = (float)mina[j];
        }
    }
}

// ---------------------------------------------------------------------------
// Kernel 3: gather z_q, commitment partial sums, segment sums/counts.
// ---------------------------------------------------------------------------
__global__ __launch_bounds__(256) void vq_gather_kernel(
    const float* __restrict__ z_e, const float* __restrict__ emb,
    const int* __restrict__ codeI, float* __restrict__ zq_out,
    float* __restrict__ counts, float* __restrict__ sums,
    float* __restrict__ commitAcc)
{
    const int row = blockIdx.x;
    const int d   = threadIdx.x;
    const int c   = codeI[row];
    const float z = z_e[(size_t)row * D + d];
    const float e = emb[(size_t)c * D + d];
    zq_out[(size_t)row * D + d] = e;   // z_q_st == z_q in value
    atomicAdd(&sums[(size_t)c * D + d], z);
    const float diff = z - e;
    __shared__ float s[256];
    s[d] = diff * diff;
    __syncthreads();
    #pragma unroll
    for (int o = 128; o > 0; o >>= 1) {
        if (d < o) s[d] += s[d + o];
        __syncthreads();
    }
    if (d == 0) {
        atomicAdd(commitAcc, s[0]);
        atomicAdd(&counts[c], 1.0f);
    }
}

// ---------------------------------------------------------------------------
// Kernel 4a: cluster_size EMA + global n reduction.
// ---------------------------------------------------------------------------
__global__ __launch_bounds__(256) void vq_size_kernel(
    const float* __restrict__ cluster_size, const float* __restrict__ counts,
    float* __restrict__ csUpd, float* __restrict__ nAcc)
{
    const int k = blockIdx.x * 256 + threadIdx.x;
    const float v = cluster_size[k] * 0.99f + 0.01f * counts[k];
    csUpd[k] = v;
    __shared__ float s[256];
    s[threadIdx.x] = v;
    __syncthreads();
    #pragma unroll
    for (int o = 128; o > 0; o >>= 1) {
        if (threadIdx.x < o) s[threadIdx.x] += s[threadIdx.x + o];
        __syncthreads();
    }
    if (threadIdx.x == 0) atomicAdd(nAcc, s[0]);
}

// ---------------------------------------------------------------------------
// Kernel 4b: cluster_sum EMA, smoothed embedding update, commitment scalar.
// ---------------------------------------------------------------------------
__global__ __launch_bounds__(256) void vq_update_kernel(
    const float* __restrict__ cluster_sum, const float* __restrict__ sums,
    const float* __restrict__ csUpd, const float* __restrict__ nAcc,
    const float* __restrict__ commitAcc, float* __restrict__ csumUpd,
    float* __restrict__ embUpd, float* __restrict__ commitOut)
{
    const int idx = blockIdx.x * 256 + threadIdx.x;
    const int k   = idx >> 8;
    const float su = cluster_sum[idx] * 0.99f + 0.01f * sums[idx];
    csumUpd[idx] = su;
    const float n = *nAcc;
    const float smoothed = (csUpd[k] + 1e-5f) / (n + (float)K * 1e-5f) * n;
    embUpd[idx] = su / smoothed;
    if (idx == 0) *commitOut = (*commitAcc) * 0.25f / (float)NELEM;
}

// ---------------------------------------------------------------------------
extern "C" void kernel_launch(void* const* d_in, const int* in_sizes, int n_in,
                              void* d_out, int out_size, void* d_ws, size_t ws_size,
                              hipStream_t stream)
{
    const float* z_e          = (const float*)d_in[0];
    const float* embedding    = (const float*)d_in[1];
    const float* cluster_size = (const float*)d_in[2];
    const float* cluster_sum  = (const float*)d_in[3];

    // Output layout (floats, reference return order):
    float* o_zq     = (float*)d_out;                 // 16,777,216
    float* o_code   = o_zq + NELEM;                  // 65,536
    float* o_commit = o_code + NROWS;                // 1
    float* o_embU   = o_commit + 1;                  // 1,048,576
    float* o_csU    = o_embU + (size_t)K * D;        // 4,096
    float* o_csumU  = o_csU + K;                     // 1,048,576

    // Workspace layout (bytes):
    char* ws = (char*)d_ws;
    _Float16* w_embF16 = (_Float16*)(ws);                         // 2,097,152 B
    float*    w_enorm  = (float*)(ws + 2097152);                  //    16,384 B
    int*      w_codeI  = (int*)(ws + 2113536);                    //   262,144 B
    float*    w_counts = (float*)(ws + 2375680);                  //    16,384 B
    float*    w_sums   = (float*)(ws + 2392064);                  // 4,194,304 B
    float*    w_nAcc   = (float*)(ws + 6586368);                  //         4 B
    float*    w_commit = (float*)(ws + 6586372);                  //         4 B

    // Zero the accumulator region (counts | sums | nAcc | commitAcc).
    (void)hipMemsetAsync(ws + 2375680, 0, 4210696, stream);

    vq_prep_kernel<<<K, 256, 0, stream>>>(embedding, w_embF16, w_enorm);

    vq_argmin_kernel<<<NROWS / 128, 256, 0, stream>>>(
        z_e, w_embF16, w_enorm, w_codeI, o_code);

    vq_gather_kernel<<<NROWS, 256, 0, stream>>>(
        z_e, embedding, w_codeI, o_zq, w_counts, w_sums, w_commit);

    vq_size_kernel<<<K / 256, 256, 0, stream>>>(
        cluster_size, w_counts, o_csU, w_nAcc);

    vq_update_kernel<<<(K * D) / 256, 256, 0, stream>>>(
        cluster_sum, w_sums, o_csU, w_nAcc, w_commit,
        o_csumU, o_embU, o_commit);
}
